// RelNet_15736760172578
// MI455X (gfx1250) — matrix operand order repaired and checked by reading, not compiled, not run on hardware
//
#include <hip/hip_runtime.h>
#include <hip/hip_bf16.h>

typedef __attribute__((ext_vector_type(16))) _Float16 v16h;
typedef __attribute__((ext_vector_type(8)))  _Float16 v8h;
typedef __attribute__((ext_vector_type(8)))  float    v8f;
typedef __attribute__((ext_vector_type(4)))  unsigned int v4u;

#define RN_N   64
#define RN_HW  64
#define RN_G   512
#define MT     128          // pair-rows per workgroup tile
#define NTHR   512          // 16 waves (wave32)

// ---------------------------------------------------------------------------
// Prep: P1[n*64+i][o] = feat(n,i) @ W1a ; P2 = feat @ W1b  (feat = [img | row | col])
// ---------------------------------------------------------------------------
__global__ void rn_prep_p(const float* __restrict__ img, const float* __restrict__ gw1,
                          float* __restrict__ P1, float* __restrict__ P2) {
  int idx = blockIdx.x * blockDim.x + threadIdx.x;      // 4096*512
  if (idx >= RN_N * RN_HW * RN_G) return;
  int o   = idx & 511;
  int row = idx >> 9;                                   // n*64 + i
  int n = row >> 6, i = row & 63;
  const float* ib = img + ((long)n << 12) + i;          // img[n][c][i], stride 64
  float a = 0.f, b = 0.f;
  #pragma unroll 4
  for (int c = 0; c < 64; ++c) {
    float f = ib[c << 6];
    a += f * gw1[c * 512 + o];
    b += f * gw1[(66 + c) * 512 + o];
  }
  float fr = (float)(i >> 3), fc = (float)(i & 7);
  a += fr * gw1[64 * 512 + o] + fc * gw1[65 * 512 + o];
  b += fr * gw1[(66 + 64) * 512 + o] + fc * gw1[(66 + 65) * 512 + o];
  P1[idx] = a;
  P2[idx] = b;
}

// Qc[n][o] = b1[o] + ques(n) @ W1c
__global__ void rn_prep_q(const float* __restrict__ ques, const float* __restrict__ gw1,
                          const float* __restrict__ gb1, float* __restrict__ Qc) {
  int idx = blockIdx.x * blockDim.x + threadIdx.x;      // 64*512
  if (idx >= RN_N * RN_G) return;
  int o = idx & 511, n = idx >> 9;
  float s = gb1[o];
  const float* q = ques + n * 256;
  #pragma unroll 4
  for (int e = 0; e < 256; ++e) s += q[e] * gw1[(132 + e) * 512 + o];
  Qc[idx] = s;
}

// Transpose+convert g_w{2,3,4} (K,N f32 row-major) -> N-major f16 [n][k]
__global__ void rn_conv_w(const float* __restrict__ w2, const float* __restrict__ w3,
                          const float* __restrict__ w4, _Float16* __restrict__ t2,
                          _Float16* __restrict__ t3, _Float16* __restrict__ t4) {
  int idx = blockIdx.x * blockDim.x + threadIdx.x;      // 512*512
  if (idx >= RN_G * RN_G) return;
  int n = idx >> 9, k = idx & 511;
  t2[idx] = (_Float16)w2[k * 512 + n];
  t3[idx] = (_Float16)w3[k * 512 + n];
  t4[idx] = (_Float16)w4[k * 512 + n];
}

__global__ void rn_zero(float* __restrict__ p, int nElem) {
  int i = blockIdx.x * blockDim.x + threadIdx.x;
  if (i < nElem) p[i] = 0.f;
}

// ---------------------------------------------------------------------------
// Fused g-MLP: layer1 reconstruct + 3x (512x512 f16 WMMA GEMM + bias + relu)
// + pair-mean accumulation into ctx via f32 atomics. One WG = 128 pair rows.
// ---------------------------------------------------------------------------
__global__ __launch_bounds__(NTHR, 1)
void rn_fused(const float* __restrict__ P1, const float* __restrict__ P2,
              const float* __restrict__ Qc,
              const _Float16* __restrict__ Wt2, const _Float16* __restrict__ Wt3,
              const _Float16* __restrict__ Wt4,
              const float* __restrict__ b2, const float* __restrict__ b3,
              const float* __restrict__ b4, float* __restrict__ ctx) {
  __shared__ __align__(32) _Float16 Xa[MT][RN_G];       // 128 KB activations
  __shared__ __align__(16) _Float16 Ws[RN_G][32];       //  32 KB weight slab (N-major)

  const int tid  = threadIdx.x;
  const int lane = tid & 31;
  const int wid  = tid >> 5;              // 0..15
  const int rs   = wid & 7;               // 16-row strip
  const int hh   = wid >> 3;              // 256-column half

  const long Rbase = (long)blockIdx.x * MT;
  const int  n_img = (int)(Rbase >> 12);  // 4096 pairs per image; tile stays in one image

  // ---- stage 0: layer-1 activations relu(P1[i]+P2[j]+Qc) into LDS as f16 ----
  {
    const float* qr = Qc + ((long)n_img << 9);
    float qv = qr[tid];
    for (int m = 0; m < MT; ++m) {
      int pair = (int)((Rbase + m) & 4095);
      int i = pair >> 6, j = pair & 63;
      const float* p1r = P1 + ((long)(n_img * 64 + i) << 9);
      const float* p2r = P2 + ((long)(n_img * 64 + j) << 9);
      float v = p1r[tid] + p2r[tid] + qv;
      Xa[m][tid] = (_Float16)(v > 0.f ? v : 0.f);
    }
  }

  const _Float16* Wts[3] = {Wt2, Wt3, Wt4};
  const float*    bss[3] = {b2, b3, b4};

  const int arow  = rs * 16 + (lane & 15);
  const int akoff = (lane >> 4) * 16;     // A-frag K sub-offset (contiguous k = 16h + s)
  const int bl    = lane & 15;            // B-frag N lane
  const int bko   = (lane >> 4) * 16;     // B-frag K sub-offset

  #pragma unroll 1
  for (int layer = 0; layer < 3; ++layer) {
    const _Float16* Wt   = Wts[layer];
    const float*    bias = bss[layer];

    v8f acc[16] = {};

    #pragma unroll 1
    for (int k0 = 0; k0 < RN_G; k0 += 32) {
      __syncthreads();                    // slab consumed / Xa writes visible
      {                                   // cooperative slab load: row n = tid, 32 B
        const v4u* src = (const v4u*)(Wt + ((long)tid << 9) + k0);
        v4u a0 = src[0], a1 = src[1];
        *(v4u*)&Ws[tid][0] = a0;
        *(v4u*)&Ws[tid][8] = a1;
        if (k0 + 32 < RN_G)
          __builtin_prefetch(Wt + ((long)tid << 9) + k0 + 32, 0, 1);
      }
      __syncthreads();

      union { v16h v; v8h h[2]; } af;     // A fragment from LDS
      af.h[0] = *(const v8h*)&Xa[arow][k0 + akoff];
      af.h[1] = *(const v8h*)&Xa[arow][k0 + akoff + 8];

      #pragma unroll
      for (int t = 0; t < 16; ++t) {
        int n = hh * 256 + t * 16 + bl;
        v16h bf = *(const v16h*)&Ws[n][bko];
        acc[t] = __builtin_amdgcn_wmma_f32_16x16x32_f16(
            false, af.v, false, bf, (short)0, acc[t], false, false);
      }
    }
    __syncthreads();                      // all reads of Xa for this layer done

    if (layer < 2) {                      // bias + relu + store back to Xa (f16)
      int rowb = rs * 16 + (lane >> 4) * 8;
      #pragma unroll
      for (int t = 0; t < 16; ++t) {
        int n = hh * 256 + t * 16 + bl;
        float b = bias[n];
        #pragma unroll
        for (int g = 0; g < 8; ++g) {
          float v = acc[t][g] + b;
          Xa[rowb + g][n] = (_Float16)(v > 0.f ? v : 0.f);
        }
      }
    } else {                              // layer 4: column-sum over M, atomic into ctx
      #pragma unroll
      for (int t = 0; t < 16; ++t) {
        int n = hh * 256 + t * 16 + bl;
        float b = bias[n];
        float s = 0.f;
        #pragma unroll
        for (int g = 0; g < 8; ++g) {
          float v = acc[t][g] + b;
          s += (v > 0.f ? v : 0.f);
        }
        atomicAdd(&ctx[((long)n_img << 9) + n], s);
      }
    }
  }
}

// ---------------------------------------------------------------------------
// f-MLP head (tiny): dense relu layers + 2-class log_softmax
// ---------------------------------------------------------------------------
__global__ void rn_head_lin(const float* __restrict__ x, const float* __restrict__ w,
                            const float* __restrict__ b, float* __restrict__ y,
                            float scale) {
  int idx = blockIdx.x * blockDim.x + threadIdx.x;      // 64*512
  if (idx >= RN_N * RN_G) return;
  int o = idx & 511, n = idx >> 9;
  const float* xr = x + ((long)n << 9);
  float s = 0.f;
  #pragma unroll 4
  for (int k = 0; k < 512; ++k) s += xr[k] * w[k * 512 + o];
  s = s * scale + b[o];
  y[idx] = s > 0.f ? s : 0.f;
}

__global__ void rn_head_out(const float* __restrict__ z2, const float* __restrict__ w3,
                            const float* __restrict__ b3, float* __restrict__ out) {
  __shared__ float sc[RN_N][2];
  int tid = threadIdx.x;
  if (tid < 128) {
    int n = tid >> 1, cls = tid & 1;
    const float* z = z2 + ((long)n << 9);
    float s = b3[cls];
    #pragma unroll 4
    for (int k = 0; k < 512; ++k) s += z[k] * w3[k * 2 + cls];
    sc[n][cls] = s;
  }
  __syncthreads();
  if (tid < 128) {
    int n = tid >> 1, cls = tid & 1;
    float a = sc[n][0], bb = sc[n][1];
    float m = fmaxf(a, bb);
    float lse = m + logf(expf(a - m) + expf(bb - m));
    out[n * 2 + cls] = sc[n][cls] - lse;
  }
}

// ---------------------------------------------------------------------------
extern "C" void kernel_launch(void* const* d_in, const int* in_sizes, int n_in,
                              void* d_out, int out_size, void* d_ws, size_t ws_size,
                              hipStream_t stream) {
  const float* img  = (const float*)d_in[0];
  const float* ques = (const float*)d_in[1];
  const float* gw1  = (const float*)d_in[2];
  const float* gb1  = (const float*)d_in[3];
  const float* gw2  = (const float*)d_in[4];
  const float* gb2  = (const float*)d_in[5];
  const float* gw3  = (const float*)d_in[6];
  const float* gb3  = (const float*)d_in[7];
  const float* gw4  = (const float*)d_in[8];
  const float* gb4  = (const float*)d_in[9];
  const float* fw1  = (const float*)d_in[10];
  const float* fb1  = (const float*)d_in[11];
  const float* fw2  = (const float*)d_in[12];
  const float* fb2  = (const float*)d_in[13];
  const float* fw3  = (const float*)d_in[14];
  const float* fb3  = (const float*)d_in[15];
  float* out = (float*)d_out;

  char* ws = (char*)d_ws;
  const size_t SZ_P  = (size_t)RN_N * RN_HW * RN_G * 4;   // 8 MB
  const size_t SZ_Q  = (size_t)RN_N * RN_G * 4;           // 128 KB
  const size_t SZ_W  = (size_t)RN_G * RN_G * 2;           // 512 KB f16
  float*    P1  = (float*)(ws);
  float*    P2  = (float*)(ws + SZ_P);
  float*    Qc  = (float*)(ws + 2 * SZ_P);
  _Float16* Wt2 = (_Float16*)(ws + 2 * SZ_P + SZ_Q);
  _Float16* Wt3 = (_Float16*)(ws + 2 * SZ_P + SZ_Q + SZ_W);
  _Float16* Wt4 = (_Float16*)(ws + 2 * SZ_P + SZ_Q + 2 * SZ_W);
  float*    ctx = (float*)(ws + 2 * SZ_P + SZ_Q + 3 * SZ_W);
  float*    z1  = (float*)(ws + 2 * SZ_P + SZ_Q + 3 * SZ_W + SZ_Q);
  float*    z2  = (float*)(ws + 2 * SZ_P + SZ_Q + 3 * SZ_W + 2 * SZ_Q);

  rn_conv_w<<<(RN_G * RN_G + 255) / 256, 256, 0, stream>>>(gw2, gw3, gw4, Wt2, Wt3, Wt4);
  rn_prep_p<<<(RN_N * RN_HW * RN_G + 255) / 256, 256, 0, stream>>>(img, gw1, P1, P2);
  rn_prep_q<<<(RN_N * RN_G + 255) / 256, 256, 0, stream>>>(ques, gw1, gb1, Qc);
  rn_zero<<<(RN_N * RN_G + 255) / 256, 256, 0, stream>>>(ctx, RN_N * RN_G);

  const int nTiles = (RN_N * RN_HW * RN_HW) / MT;         // 2048
  rn_fused<<<nTiles, NTHR, 0, stream>>>(P1, P2, Qc, Wt2, Wt3, Wt4, gb2, gb3, gb4, ctx);

  rn_head_lin<<<(RN_N * RN_G + 255) / 256, 256, 0, stream>>>(ctx, fw1, fb1, z1,
                                                             1.0f / 4096.0f);
  rn_head_lin<<<(RN_N * RN_G + 255) / 256, 256, 0, stream>>>(z1, fw2, fb2, z2, 1.0f);
  rn_head_out<<<1, 128, 0, stream>>>(z2, fw3, fb3, out);
  (void)in_sizes; (void)n_in; (void)out_size; (void)ws_size;
}
